// MHRSQ_52029233823802
// MI455X (gfx1250) — matrix operand order repaired and checked by reading, not compiled, not run on hardware
//
#include <hip/hip_runtime.h>
#include <hip/hip_bf16.h>

// ---------------------------------------------------------------------------
// CDNA5 (gfx1250) WMMA types
// ---------------------------------------------------------------------------
typedef __attribute__((ext_vector_type(16))) _Float16 v16h;
typedef __attribute__((ext_vector_type(8)))  float    v8f;
typedef __attribute__((ext_vector_type(4)))  float    f4; // 16-byte POD load unit
typedef unsigned int u32x4 __attribute__((ext_vector_type(4)));
typedef int          i32x4 __attribute__((ext_vector_type(4)));
typedef int          i32x8 __attribute__((ext_vector_type(8)));

union V16 {
    v16h     v;
    _Float16 h[16];
    f4       q[2];
};

// D = A*B + C, 16x16x32 f16 inputs, f32 accumulate
__device__ __forceinline__ v8f wmma_f16(v16h a, v16h b, v8f c) {
    return __builtin_amdgcn_wmma_f32_16x16x32_f16(
        /*neg_a=*/false, a, /*neg_b=*/false, b,
        /*c_mod=*/(short)0, c, /*reuse_a=*/false, /*reuse_b=*/false);
}

// A-matrix 16x32 f16 fragment from row-major [16 x ld] tile at `base`.
// lanes 0-15 -> M=lane, halves[0:8]=K 0..7, halves[8:16]=K16..23
// lanes16-31 -> M=lane-16, halves[0:8]=K 8..15, halves[8:16]=K24..31
__device__ __forceinline__ v16h load_a16x32(const _Float16* base, int ld, int k0, int lane) {
    int half = lane >> 4;
    int m    = lane & 15;
    const _Float16* p = base + (size_t)m * ld + k0 + half * 8;
    V16 u;
    u.q[0] = *(const f4*)(p);
    u.q[1] = *(const f4*)(p + 16);
    return u.v;
}

// B-matrix 32x16 f16 fragment. Column n of B is a contiguous row at base + n*ld.
// Lanes 0-15: column n, K 0..7,16..23; lanes 16-31: column n-16, K 8..15,24..31.
__device__ __forceinline__ v16h load_b32x16(const _Float16* base, int ld, int k0, int lane) {
    int half = lane >> 4;
    int n    = lane & 15;
    const _Float16* p = base + (size_t)n * ld + k0 + half * 8;
    V16 u;
    u.q[0] = *(const f4*)(p);
    u.q[1] = *(const f4*)(p + 16);
    return u.v;
}

// ---------------------------------------------------------------------------
// CDNA5 async global->LDS (ASYNCcnt) and TDM (TENSORcnt) helpers
// ---------------------------------------------------------------------------
__device__ __forceinline__ void async_g2l_b128(unsigned lds_byte_addr, const void* gaddr) {
    asm volatile("global_load_async_to_lds_b128 %0, %1, off"
                 :: "v"(lds_byte_addr), "v"((unsigned long long)(size_t)gaddr)
                 : "memory");
}
__device__ __forceinline__ void wait_asynccnt0() {
    asm volatile("s_wait_asynccnt 0x0" ::: "memory");
}

// 2-D TDM load: tile (tile0 x tile1) of 2-byte elements from a row-major
// tensor with row stride `stride0` elements, into LDS at byte addr `lds_addr`
// packed with row pitch tile0*2 bytes. D# bit layout per CDNA5 ISA §8.3/8.4.
__device__ __forceinline__ void tdm_load_2d(unsigned lds_addr, const void* gaddr,
                                            unsigned dim0, unsigned dim1,
                                            unsigned tile0, unsigned tile1,
                                            unsigned stride0) {
    unsigned long long ga = (unsigned long long)(size_t)gaddr;
    u32x4 g0;
    g0.x = 1u;                                                 // count=1 (valid), user mode
    g0.y = lds_addr;                                           // lds_addr [63:32]
    g0.z = (unsigned)ga;                                       // global_addr[31:0]
    g0.w = (unsigned)((ga >> 32) & 0x1FFFFFFu) | (2u << 30);   // addr[56:32] | type=2
    i32x8 g1;
    g1[0] = (int)(1u << 16);                                   // data_size=1 -> 2 bytes
    g1[1] = (int)((dim0 & 0xFFFFu) << 16);                     // tensor_dim0[15:0]
    g1[2] = (int)(((dim0 >> 16) & 0xFFFFu) | ((dim1 & 0xFFFFu) << 16));
    g1[3] = (int)(((dim1 >> 16) & 0xFFFFu) | ((tile0 & 0xFFFFu) << 16));
    g1[4] = (int)(tile1 & 0xFFFFu);                            // tile_dim1, tile_dim2=0
    g1[5] = (int)stride0;                                      // tensor_dim0_stride[31:0]
    g1[6] = 0;
    g1[7] = 0;
    i32x4 z4 = {0, 0, 0, 0};
#if defined(__clang_major__) && (__clang_major__ >= 23)
    i32x8 z8 = {0, 0, 0, 0, 0, 0, 0, 0};
    __builtin_amdgcn_tensor_load_to_lds(g0, g1, z4, z4, z8, 0);
#else
    __builtin_amdgcn_tensor_load_to_lds(g0, g1, z4, z4, 0);
#endif
}

// ---------------------------------------------------------------------------
// xor-reductions across the 16 lanes sharing a softmax row, via v_permlane16.
// ---------------------------------------------------------------------------
__device__ __forceinline__ float pl16(float x, unsigned s0, unsigned s1) {
#if __has_builtin(__builtin_amdgcn_permlane16)
    union { float f; unsigned u; } a, b;
    a.f = x;
    b.u = __builtin_amdgcn_permlane16(a.u, a.u, s0, s1, false, false);
    return b.f;
#else
    (void)s0; (void)s1;
    return x;
#endif
}
__device__ __forceinline__ float red_max16(float x) {
#if __has_builtin(__builtin_amdgcn_permlane16)
    x = fmaxf(x, pl16(x, 0x67452301u, 0xEFCDAB89u)); // xor 1
    x = fmaxf(x, pl16(x, 0x54761032u, 0xDCFE98BAu)); // xor 2
    x = fmaxf(x, pl16(x, 0x32107654u, 0xBA98FEDCu)); // xor 4
    x = fmaxf(x, pl16(x, 0xFEDCBA98u, 0x76543210u)); // xor 8
#else
#pragma unroll
    for (int m = 1; m < 16; m <<= 1) x = fmaxf(x, __shfl_xor(x, m, 32));
#endif
    return x;
}
__device__ __forceinline__ float red_sum16(float x) {
#if __has_builtin(__builtin_amdgcn_permlane16)
    x += pl16(x, 0x67452301u, 0xEFCDAB89u);
    x += pl16(x, 0x54761032u, 0xDCFE98BAu);
    x += pl16(x, 0x32107654u, 0xBA98FEDCu);
    x += pl16(x, 0xFEDCBA98u, 0x76543210u);
#else
#pragma unroll
    for (int m = 1; m < 16; m <<= 1) x += __shfl_xor(x, m, 32);
#endif
    return x;
}

// ---------------------------------------------------------------------------
// Problem constants
// ---------------------------------------------------------------------------
#define BATCH 8
#define SEQ   1024
#define DIM   768
#define HEADS 12
#define DHEAD 64
#define BH    (BATCH * HEADS)          // 96
#define ROWS  (BATCH * SEQ)            // 8192
#define NQKV  (3 * DIM)                // 2304

// ---------------------------------------------------------------------------
// 1) f32 -> f16 convert
// ---------------------------------------------------------------------------
__global__ __launch_bounds__(256) void cvt_f32_f16(const float* __restrict__ src,
                                                   _Float16* __restrict__ dst, int n) {
    int i      = blockIdx.x * blockDim.x + threadIdx.x;
    int stride = gridDim.x * blockDim.x;
    for (; i < n; i += stride) dst[i] = (_Float16)src[i];
}

// ---------------------------------------------------------------------------
// 2/5) WMMA GEMM: C[M,N] = A[M,K] * B^T (B row-major [N,K]).
// grid.x = M/64, grid.y = N/64, block = 128 (4 waves, 16 M-rows each).
// Double-buffered B-tile staging (4 KB each) via global_load_async_to_lds_b128:
// issue next k-step's DMA, compute on current, one s_wait_asynccnt + barrier.
// ---------------------------------------------------------------------------
template <bool F32OUT>
__global__ __launch_bounds__(128) void gemm_wmma(const _Float16* __restrict__ A,
                                                 const _Float16* __restrict__ B,
                                                 const float* __restrict__ bias,
                                                 void* __restrict__ Cv,
                                                 int N, int K) {
    __shared__ __align__(16) _Float16 Bt[2][64 * 32];   // [buf][n(64)][k(32)], ld=32
    const int tid  = threadIdx.x;
    const int lane = tid & 31;
    const int wv   = tid >> 5;
    const int m0   = blockIdx.x * 64 + wv * 16;
    const int n0   = blockIdx.y * 64;
    const _Float16* Arow   = A + (size_t)m0 * K;
    const unsigned  btBase = (unsigned)(size_t)&Bt[0][0];

    // stage one 64x32 B tile (256 x 16B chunks; 2 async loads per thread)
    auto stage = [&](int k0, int buf) {
#pragma unroll
        for (int c = 0; c < 2; ++c) {
            int chunk = tid + c * 128;
            int row   = chunk >> 2;          // 0..63
            int seg   = chunk & 3;           // 16-B segment within 64-B row
            const _Float16* g = B + (size_t)(n0 + row) * K + k0 + seg * 8;
            async_g2l_b128(btBase + (unsigned)(buf * 4096 + row * 64 + seg * 16), g);
        }
    };

    v8f acc[4];
#pragma unroll
    for (int t = 0; t < 4; ++t)
#pragma unroll
        for (int r = 0; r < 8; ++r) acc[t][r] = 0.0f;

    stage(0, 0);
    wait_asynccnt0();
    __syncthreads();

    for (int k0 = 0; k0 < K; k0 += 32) {
        const int cur = (k0 >> 5) & 1;
        if (k0 + 32 < K) stage(k0 + 32, cur ^ 1);   // overlap DMA with compute

        v16h a = load_a16x32(Arow, K, k0, lane);
#pragma unroll
        for (int t = 0; t < 4; ++t) {
            v16h b = load_b32x16(&Bt[cur][(16 * t) * 32], 32, 0, lane);
            acc[t] = wmma_f16(a, b, acc[t]);
        }
        wait_asynccnt0();   // next buffer landed (per-wave ASYNCcnt)
        __syncthreads();    // all waves done reading cur before it is re-staged
    }

    const int rbase = (lane >> 4) * 8;   // D frag: VGPR r -> M = r + 8*(lane/16)
    const int cidx  = lane & 15;         //        N = lane % 16
#pragma unroll
    for (int t = 0; t < 4; ++t) {
#pragma unroll
        for (int r = 0; r < 8; ++r) {
            int m = m0 + rbase + r;
            int n = n0 + 16 * t + cidx;
            if (F32OUT) {
                ((float*)Cv)[(size_t)m * N + n] = acc[t][r] + bias[n];
            } else {
                ((_Float16*)Cv)[(size_t)m * N + n] = (_Float16)acc[t][r];
            }
        }
    }
}

// ---------------------------------------------------------------------------
// 3) Rearrange qkv [ROWS, 2304] f16 ->
//    sq [bh, n, d] (perm-gathered Q), kb [bh, n, d], vt [bh, d, n]
// ---------------------------------------------------------------------------
__global__ __launch_bounds__(256) void rearrange_qkv(const _Float16* __restrict__ qkv,
                                                     const int* __restrict__ perms,
                                                     _Float16* __restrict__ sq,
                                                     _Float16* __restrict__ kb,
                                                     _Float16* __restrict__ vt) {
    int idx = blockIdx.x * 256 + threadIdx.x;   // over BH*SEQ*DHEAD = 6291456
    int d   = idx & 63;
    int n   = (idx >> 6) & 1023;
    int bh  = idx >> 16;                        // 0..95
    int h   = bh % HEADS;
    int b   = bh / HEADS;
    size_t qrow = ((size_t)b * SEQ + n) * NQKV;

    kb[idx] = qkv[qrow + DIM + h * DHEAD + d];
    vt[((size_t)bh * DHEAD + d) * SEQ + n] = qkv[qrow + 2 * DIM + h * DHEAD + d];

    int i = n * DHEAD + d;
    int p = perms[h * (SEQ * DHEAD) + i];
    sq[(size_t)bh * (SEQ * DHEAD) + i] =
        qkv[((size_t)b * SEQ + (p >> 6)) * NQKV + h * DHEAD + (p & 63)];
}

// ---------------------------------------------------------------------------
// 4) Flash attention: per (b,h, 64-row block); 4 waves, 16 Q-rows each.
//    K (32x64) and V^T (64x32) tiles double-buffered in LDS, staged by the
//    TDM (wave0 -> K, wave1 -> V^T): issue DMA for tile j+32 at the top of
//    iteration j, compute on tile j, single s_wait_tensorcnt + barrier.
// ---------------------------------------------------------------------------
__global__ __launch_bounds__(128) void attn_wmma(const _Float16* __restrict__ SQ,
                                                 const _Float16* __restrict__ KB,
                                                 const _Float16* __restrict__ VT,
                                                 _Float16* __restrict__ OH) {
    __shared__ __align__(16) _Float16 Kt[2][32 * 64];   // [buf][key(32)][k(64)], ld=64
    __shared__ __align__(16) _Float16 Vl[2][64 * 32];   // [buf][d(64)][key(32)], ld=32
    __shared__ __align__(16) _Float16 Plds[4][16 * 32]; // per-wave P tile

    const int lane = threadIdx.x & 31;
    const int wv   = threadIdx.x >> 5;
    const int mblk = blockIdx.x;        // 0..15
    const int bh   = blockIdx.y;        // 0..95
    const int b    = bh / HEADS;
    const int h    = bh % HEADS;

    const _Float16* Qb = SQ + (size_t)bh * (SEQ * DHEAD) + (size_t)(mblk * 64 + wv * 16) * DHEAD;
    const _Float16* Kb = KB + (size_t)bh * (SEQ * DHEAD);
    const _Float16* Vb = VT + (size_t)bh * (SEQ * DHEAD);
    const unsigned ktA = (unsigned)(size_t)&Kt[0][0];
    const unsigned vlA = (unsigned)(size_t)&Vl[0][0];

    const v16h a0 = load_a16x32(Qb, DHEAD, 0, lane);
    const v16h a1 = load_a16x32(Qb, DHEAD, 32, lane);

    float mrow[8], lrow[8];
    v8f acc[4];
#pragma unroll
    for (int r = 0; r < 8; ++r) { mrow[r] = -1e30f; lrow[r] = 0.0f; }
#pragma unroll
    for (int t = 0; t < 4; ++t)
#pragma unroll
        for (int r = 0; r < 8; ++r) acc[t][r] = 0.0f;

    const float scale = 0.125f;   // 1/sqrt(64)
    _Float16* P       = Plds[wv];
    const int rbase   = (lane >> 4) * 8;
    const int cidx    = lane & 15;

    auto stage_kv = [&](int j, int buf) {
        if (wv == 0) {
            tdm_load_2d(ktA + (unsigned)buf * 4096, Kb + (size_t)j * DHEAD,
                        /*dim0=*/64, /*dim1=*/32, /*tile0=*/64, /*tile1=*/32,
                        /*stride0=*/64);
        } else if (wv == 1) {
            tdm_load_2d(vlA + (unsigned)buf * 4096, Vb + j,
                        /*dim0=*/32, /*dim1=*/64, /*tile0=*/32, /*tile1=*/64,
                        /*stride0=*/SEQ);
        }
    };

    // prologue: stage tile j=0 into buffer 0
    stage_kv(0, 0);
    __builtin_amdgcn_s_wait_tensorcnt(0);
    __syncthreads();

    for (int j = 0; j < SEQ; j += 32) {
        const int cur = (j >> 5) & 1;
        if (j + 32 < SEQ) stage_kv(j + 32, cur ^ 1);   // overlap TDM with compute

        const _Float16* Kc = &Kt[cur][0];
        const _Float16* Vc = &Vl[cur][0];

        // group the 4 K B-fragment loads, then both S WMMAs
        v16h bk00 = load_b32x16(Kc, 64, 0, lane);
        v16h bk01 = load_b32x16(Kc, 64, 32, lane);
        v16h bk10 = load_b32x16(Kc + 16 * 64, 64, 0, lane);
        v16h bk11 = load_b32x16(Kc + 16 * 64, 64, 32, lane);
        v8f z = {0.f, 0.f, 0.f, 0.f, 0.f, 0.f, 0.f, 0.f};
        v8f s0 = wmma_f16(a1, bk01, wmma_f16(a0, bk00, z));
        v8f s1 = wmma_f16(a1, bk11, wmma_f16(a0, bk10, z));

        float p0[8], p1[8], alpha[8];
#pragma unroll
        for (int r = 0; r < 8; ++r) {
            float v0 = s0[r] * scale;
            float v1 = s1[r] * scale;
            float tm = red_max16(fmaxf(v0, v1));
            float mn = fmaxf(mrow[r], tm);
            float al = __expf(mrow[r] - mn);
            float e0 = __expf(v0 - mn);
            float e1 = __expf(v1 - mn);
            float rs = red_sum16(e0 + e1);
            lrow[r]  = lrow[r] * al + rs;
            mrow[r]  = mn;
            alpha[r] = al;
            p0[r]    = e0;
            p1[r]    = e1;
        }
#pragma unroll
        for (int t = 0; t < 4; ++t)
#pragma unroll
            for (int r = 0; r < 8; ++r) acc[t][r] *= alpha[r];

        // D-frag (col-per-lane) -> LDS row-major [16 x 32] f16
#pragma unroll
        for (int r = 0; r < 8; ++r) {
            P[(rbase + r) * 32 + cidx]      = (_Float16)p0[r];
            P[(rbase + r) * 32 + 16 + cidx] = (_Float16)p1[r];
        }
        // Re-load as A-frag (row-per-lane); same-wave LDS ops are in order.
        v16h pa  = load_a16x32(P, 32, 0, lane);
        // group the 4 V^T B-fragment loads, then the 4 PV WMMAs
        v16h bv0 = load_b32x16(Vc + 0 * 32, 32, 0, lane);
        v16h bv1 = load_b32x16(Vc + 16 * 32, 32, 0, lane);
        v16h bv2 = load_b32x16(Vc + 32 * 32, 32, 0, lane);
        v16h bv3 = load_b32x16(Vc + 48 * 32, 32, 0, lane);
        acc[0] = wmma_f16(pa, bv0, acc[0]);
        acc[1] = wmma_f16(pa, bv1, acc[1]);
        acc[2] = wmma_f16(pa, bv2, acc[2]);
        acc[3] = wmma_f16(pa, bv3, acc[3]);

        __builtin_amdgcn_s_wait_tensorcnt(0);   // next buffer landed (waves 0/1)
        __syncthreads();                        // everyone done with cur buffer
    }

    // Epilogue: O[b, m, h*64 + d] = acc / l   (f16 into [ROWS, 768])
    const int mbase = mblk * 64 + wv * 16 + rbase;
#pragma unroll
    for (int r = 0; r < 8; ++r) {
        float inv = 1.0f / lrow[r];
        int   m   = mbase + r;
        size_t ro = ((size_t)b * SEQ + m) * DIM + h * DHEAD;
#pragma unroll
        for (int t = 0; t < 4; ++t)
            OH[ro + 16 * t + cidx] = (_Float16)(acc[t][r] * inv);
    }
}

// ---------------------------------------------------------------------------
// Launch
// ---------------------------------------------------------------------------
extern "C" void kernel_launch(void* const* d_in, const int* in_sizes, int n_in,
                              void* d_out, int out_size, void* d_ws, size_t ws_size,
                              hipStream_t stream) {
    const float* x     = (const float*)d_in[0];   // [8,1024,768]
    const float* w_qkv = (const float*)d_in[1];   // [2304,768]
    const float* w_out = (const float*)d_in[2];   // [768,768]
    const float* b_out = (const float*)d_in[3];   // [768]
    const int*   perms = (const int*)d_in[4];     // [12, 65536]
    float*       out   = (float*)d_out;           // [8,1024,768]

    _Float16* ws  = (_Float16*)d_ws;
    size_t    off = 0;
    _Float16* xh    = ws + off; off += (size_t)ROWS * DIM;       // reused as oh
    _Float16* wqkvh = ws + off; off += (size_t)NQKV * DIM;
    _Float16* wouth = ws + off; off += (size_t)DIM * DIM;
    _Float16* qkvh  = ws + off; off += (size_t)ROWS * NQKV;
    _Float16* sq    = ws + off; off += (size_t)BH * SEQ * DHEAD;
    _Float16* kb    = ws + off; off += (size_t)BH * SEQ * DHEAD;
    _Float16* vt    = ws + off; off += (size_t)BH * SEQ * DHEAD;
    _Float16* oh    = xh; // xh is dead after GEMM1

    // 1) convert inputs to f16
    cvt_f32_f16<<<1024, 256, 0, stream>>>(x, xh, ROWS * DIM);
    cvt_f32_f16<<<512, 256, 0, stream>>>(w_qkv, wqkvh, NQKV * DIM);
    cvt_f32_f16<<<512, 256, 0, stream>>>(w_out, wouth, DIM * DIM);

    // 2) qkv = x @ w_qkv^T   (8192 x 2304 x 768)
    gemm_wmma<false><<<dim3(ROWS / 64, NQKV / 64), 128, 0, stream>>>(
        xh, wqkvh, nullptr, qkvh, NQKV, DIM);

    // 3) split + permute-gather Q, build K and V^T per head
    rearrange_qkv<<<(BH * SEQ * DHEAD) / 256, 256, 0, stream>>>(qkvh, perms, sq, kb, vt);

    // 4) flash attention -> oh [8192, 768] f16
    attn_wmma<<<dim3(SEQ / 64, BH), 128, 0, stream>>>(sq, kb, vt, oh);

    // 5) out = oh @ w_out^T + b_out   (8192 x 768 x 768), f32
    gemm_wmma<true><<<dim3(ROWS / 64, DIM / 64), 128, 0, stream>>>(
        oh, wouth, b_out, out, DIM, DIM);
}
